// DARM_43267500540057
// MI455X (gfx1250) — matrix operand order repaired and checked by reading, not compiled, not run on hardware
//
#include <hip/hip_runtime.h>
#include <math.h>

// ---------------------------------------------------------------------------
// MI455X (gfx1250) implementation.
// Conv trunk as implicit GEMM on v_wmma_f32_16x16x32_bf16 (wave32):
//   A = 32 pixels x 32 channels (from an LDS-staged 3x34x128 halo tile)
//   B = 32 K x 16 out-channels from weights packed [O][9*128] bf16
//   C/D = two f32 16x16 accumulators per wave (M-tile = 32 pixels)
// Per-lane layouts follow CDNA5 ISA 7.12.2:
//   A: lane l (row M = l&15) holds K = c0..c0+7 and c0+16..c0+23,
//      c0 = chunk*32 + (l>=16 ? 8 : 0)   -> two 128-bit loads
//   B: lane l (col N = l&15) holds K = c0..c0+7 and c0+16..c0+23,
//      same c0 as A                      -> two 128-bit loads
//   D: reg r -> row M = r + (l>=16 ? 8 : 0), col N = l&15
// M=32 per block => each B fragment feeds TWO WMMAs (halves weight L2
// traffic; weights are fully L2-resident in the 192MB L2 anyway, features
// are the HBM term at ~67MB/pass vs 23.3 TB/s).
// Zero padding is resolved once while staging the halo tile into LDS, so the
// 72-WMMA K-loop is branch-free straight-line code (no EXEC manipulation,
// A from ds_load_b128, B from global_load_b128 clauses).
// LDS pixel stride = 136 elems (272 B): consecutive pixel rows advance 4
// banks, so the 16 A-lanes' b128 reads are conflict-free.
// ---------------------------------------------------------------------------

typedef __bf16 bf16_t;
typedef __attribute__((ext_vector_type(16))) __bf16 v16bf;
typedef __attribute__((ext_vector_type(8)))  __bf16 v8bf;
typedef __attribute__((ext_vector_type(8)))  float  v8f;

static inline int cdiv_i(long long a, long long b) { return (int)((a + b - 1) / b); }

#define TILE_PSTRIDE 136   // bf16 elements per staged pixel row (128 + 8 pad)
#define MTILE 32           // pixels per block
#define HALO (MTILE + 2)   // staged pixels per row

// ---- pack conv weights [O,I,3,3] f32 -> [Opad][9*I] bf16, K=(ky*3+kx)*I+i --
__global__ void pack_w_kernel(const float* __restrict__ w, bf16_t* __restrict__ wpk,
                              int O, int Opad, int I) {
    int Ktot = 9 * I;
    long long idx = (long long)blockIdx.x * blockDim.x + threadIdx.x;
    if (idx >= (long long)Opad * Ktot) return;
    int o = (int)(idx / Ktot), k = (int)(idx % Ktot);
    int tap = k / I, i = k % I;
    float v = 0.f;
    if (o < O) v = w[(long long)(o * I + i) * 9 + tap];
    wpk[(long long)o * Ktot + k] = (bf16_t)v;
}

// ---- head conv 3->128, pad 1, NCHW f32 in -> NHWC bf16 out (no relu) ------
__global__ void head_conv_kernel(const float* __restrict__ x,
                                 const float* __restrict__ w,
                                 const float* __restrict__ bias,
                                 bf16_t* __restrict__ fout,
                                 int B, int H, int W) {
    long long idx = (long long)blockIdx.x * blockDim.x + threadIdx.x;
    long long total = (long long)B * H * W * 128;
    if (idx >= total) return;
    int o = (int)(idx & 127);
    long long p = idx >> 7;
    int HW = H * W;
    int b = (int)(p / HW);
    int rem = (int)(p % HW);
    int y = rem / W, xx0 = rem % W;
    float s = bias[o];
    #pragma unroll
    for (int c = 0; c < 3; ++c) {
        #pragma unroll
        for (int ky = 0; ky < 3; ++ky) {
            int yy = y + ky - 1;
            if ((unsigned)yy >= (unsigned)H) continue;
            #pragma unroll
            for (int kx = 0; kx < 3; ++kx) {
                int xw = xx0 + kx - 1;
                if ((unsigned)xw >= (unsigned)W) continue;
                s = fmaf(x[((long long)(b * 3 + c) * H + yy) * W + xw],
                         w[((o * 3 + c) * 3 + ky) * 3 + kx], s);
            }
        }
    }
    fout[idx] = (bf16_t)s;
}

// ---- 3x3 conv, I=128, implicit GEMM on WMMA bf16, LDS-staged halo tile ----
// grid.x = B*H*W/32 (one 32-pixel row-tile per block); block = n_otile*32.
__global__ void conv3x3_wmma_kernel(const bf16_t* __restrict__ fin,
                                    const bf16_t* __restrict__ wpk,
                                    const float* __restrict__ bias,
                                    bf16_t* __restrict__ fout_bf,
                                    float* __restrict__ fout_f32,
                                    int H, int W, int relu, int ocnt, int ostride) {
    const int I = 128, Ktot = 9 * 128;
    __shared__ bf16_t tile[3 * HALO * TILE_PSTRIDE];   // 3 rows x 34 pixels

    const int HW = H * W;
    const int P = blockIdx.x * MTILE;  // base pixel of tile
    const int b = P / HW;
    const int rem = P - b * HW;
    const int y = rem / W;
    const int x0 = rem - y * W;        // tiles never cross rows (W%32==0)

    // ---- stage halo tile: 3*34 pixels x 16 v8bf groups; zero-pad here -----
    const v8bf z8 = {};
    for (int j = threadIdx.x; j < 3 * HALO * 16; j += blockDim.x) {
        int pix = j >> 4, sub = j & 15;          // sub: which 8-channel group
        int r = pix / HALO, t = pix - r * HALO;  // r: row 0..2, t: x 0..33
        int yy = y + r - 1, xw = x0 + t - 1;
        v8bf v = z8;
        if ((unsigned)yy < (unsigned)H && (unsigned)xw < (unsigned)W)
            v = ((const v8bf*)(fin + (long long)((b * H + yy) * W + xw) * I))[sub];
        *(v8bf*)(tile + pix * TILE_PSTRIDE + sub * 8) = v;
    }
    __syncthreads();

    const int wave = threadIdx.x >> 5;
    const int lane = threadIdx.x & 31;
    const int hl   = lane >> 4;        // which 16-lane half
    const int nrow = lane & 15;
    const int ocol = wave * 16 + nrow; // this lane's B column / store channel

    v8f acc0 = {};                     // pixels P .. P+15
    v8f acc1 = {};                     // pixels P+16 .. P+31
    const bf16_t* wrow = wpk + (long long)ocol * Ktot;

    #pragma unroll
    for (int ky = 0; ky < 3; ++ky) {
        #pragma unroll
        for (int kx = 0; kx < 3; ++kx) {
            const int tap = ky * 3 + kx;
            // A rows for this tap: LDS pixel (ky, nrow+kx); padding already 0
            const bf16_t* arow0 = tile + (ky * HALO + nrow + kx) * TILE_PSTRIDE;
            const bf16_t* arow1 = arow0 + 16 * TILE_PSTRIDE;
            if (tap < 8)  // pull next tap's weight row toward the WGP
                __builtin_prefetch(wrow + (tap + 1) * I, 0, 3);
            #pragma unroll
            for (int ch = 0; ch < 4; ++ch) {
                union { v16bf v; v8bf h[2]; } Bm;
                const v8bf* bp = (const v8bf*)(wrow + tap * I + ch * 32 + hl * 8);
                Bm.h[0] = bp[0]; Bm.h[1] = bp[2];
                union { v16bf v; v8bf h[2]; } A0, A1;
                const bf16_t* ac0 = arow0 + ch * 32 + hl * 8;
                const bf16_t* ac1 = arow1 + ch * 32 + hl * 8;
                A0.h[0] = *(const v8bf*)(ac0);
                A0.h[1] = *(const v8bf*)(ac0 + 16);
                A1.h[0] = *(const v8bf*)(ac1);
                A1.h[1] = *(const v8bf*)(ac1 + 16);
                acc0 = __builtin_amdgcn_wmma_f32_16x16x32_bf16(
                        false, A0.v, false, Bm.v, (short)0, acc0, false, false);
                acc1 = __builtin_amdgcn_wmma_f32_16x16x32_bf16(
                        false, A1.v, false, Bm.v, (short)0, acc1, false, false);
            }
        }
    }

    float bo = (ocol < ocnt) ? bias[ocol] : 0.f;
    #pragma unroll
    for (int r = 0; r < 8; ++r) {
        int m = r + hl * 8;                          // D row -> pixel offset
        float v0 = acc0[r] + bo;
        float v1 = acc1[r] + bo;
        if (relu) { v0 = fmaxf(v0, 0.f); v1 = fmaxf(v1, 0.f); }
        long long off0 = (long long)(P + m) * ostride + ocol;
        long long off1 = (long long)(P + 16 + m) * ostride + ocol;
        if (fout_bf) {
            fout_bf[off0] = (bf16_t)v0;
            fout_bf[off1] = (bf16_t)v1;
        } else if (ocol < ocnt) {
            fout_f32[off0] = v0;
            fout_f32[off1] = v1;
        }
    }
}

// ---- SE: global average pool over HW per (b,c) ----------------------------
__global__ void se_pool_kernel(const bf16_t* __restrict__ feat,
                               float* __restrict__ pooled, int HW) {
    int bc = blockIdx.x;               // b*128 + c
    int c = bc & 127, b = bc >> 7;
    const bf16_t* p = feat + (long long)b * HW * 128 + c;
    float s = 0.f;
    for (int i = threadIdx.x; i < HW; i += blockDim.x) s += (float)p[(long long)i * 128];
    __shared__ float red[256];
    red[threadIdx.x] = s;
    __syncthreads();
    for (int st = 128; st > 0; st >>= 1) {
        if ((int)threadIdx.x < st) red[threadIdx.x] += red[threadIdx.x + st];
        __syncthreads();
    }
    if (threadIdx.x == 0) pooled[bc] = red[0] / (float)HW;
}

// ---- SE: 128 -> S(relu) -> 128(sigmoid); grid=B, block=128 ----------------
__global__ void se_mlp_kernel(const float* __restrict__ pooled,
                              const float* __restrict__ w1, const float* __restrict__ b1,
                              const float* __restrict__ w2, const float* __restrict__ b2,
                              float* __restrict__ scale, int S) {
    int b = blockIdx.x, t = threadIdx.x;
    __shared__ float ps[128];
    __shared__ float hid[8];
    ps[t] = pooled[b * 128 + t];
    __syncthreads();
    if (t < S) {
        float h = b1[t];
        for (int c = 0; c < 128; ++c) h = fmaf(w1[t * 128 + c], ps[c], h);
        hid[t] = fmaxf(h, 0.f);
    }
    __syncthreads();
    float sc = b2[t];
    for (int s = 0; s < S; ++s) sc = fmaf(w2[t * S + s], hid[s], sc);
    scale[b * 128 + t] = 1.f / (1.f + expf(-sc));
}

// ---- SE: f *= s (broadcast per b,c), NHWC bf16 in place -------------------
__global__ void se_scale_kernel(bf16_t* feat, const float* __restrict__ scale, int HW,
                                long long total) {
    long long idx = (long long)blockIdx.x * blockDim.x + threadIdx.x;
    if (idx >= total) return;
    int c = (int)(idx & 127);
    long long p = idx >> 7;
    int b = (int)(p / HW);
    feat[idx] = (bf16_t)((float)feat[idx] * scale[b * 128 + c]);
}

// ---- dynamic filtering: out[b,c,y,x] = sum_k wgt[b,y,x,k]*(basis_k (*) x) --
// Folds basis into one effective per-pixel 5x5 kernel first (72*25 + 3*25).
__global__ void combine_kernel(const float* __restrict__ x,
                               const float* __restrict__ wgt,
                               const float* __restrict__ basis,
                               float* __restrict__ out,
                               int B, int H, int W) {
    __shared__ float bs[72 * 25];
    for (int i = threadIdx.x; i < 72 * 25; i += blockDim.x) bs[i] = basis[i];
    __syncthreads();
    long long p = (long long)blockIdx.x * blockDim.x + threadIdx.x;  // pixel id
    int HW = H * W;
    if (p >= (long long)B * HW) return;
    int b = (int)(p / HW);
    int rem = (int)(p % HW);
    int y = rem / W, xx = rem % W;

    float eff[25];
    #pragma unroll
    for (int t = 0; t < 25; ++t) eff[t] = 0.f;
    const float* wp = wgt + p * 72;
    for (int k = 0; k < 72; ++k) {
        float wk = wp[k];
        const float* bk = bs + k * 25;
        #pragma unroll
        for (int t = 0; t < 25; ++t) eff[t] = fmaf(wk, bk[t], eff[t]);
    }
    #pragma unroll
    for (int c = 0; c < 3; ++c) {
        const float* xc = x + (long long)(b * 3 + c) * HW;
        float s = 0.f;
        #pragma unroll
        for (int dy = 0; dy < 5; ++dy) {
            int yy = y + dy - 2;
            yy = yy < 0 ? -yy : (yy >= H ? 2 * H - 2 - yy : yy);   // reflect pad
            #pragma unroll
            for (int dx = 0; dx < 5; ++dx) {
                int xw = xx + dx - 2;
                xw = xw < 0 ? -xw : (xw >= W ? 2 * W - 2 - xw : xw);
                s = fmaf(eff[dy * 5 + dx], xc[yy * W + xw], s);
            }
        }
        out[(long long)(b * 3 + c) * HW + rem] = s;
    }
}

// ---------------------------------------------------------------------------
extern "C" void kernel_launch(void* const* d_in, const int* in_sizes, int n_in,
                              void* d_out, int out_size, void* d_ws, size_t ws_size,
                              hipStream_t stream) {
    (void)in_sizes; (void)n_in; (void)out_size; (void)ws_size;
    const int B = 8, H = 128, W = 128, S = 8;
    const int HW = H * W;
    const long long NPIX = (long long)B * HW;          // 131072
    const long long FE   = NPIX * 128;                 // feature elements

    const float* x       = (const float*)d_in[0];
    const float* w_head  = (const float*)d_in[1];
    const float* b_head  = (const float*)d_in[2];
    const float* w_h1a   = (const float*)d_in[3];
    const float* b_h1a   = (const float*)d_in[4];
    const float* w_h1b   = (const float*)d_in[5];
    const float* b_h1b   = (const float*)d_in[6];
    const float* du1_w1  = (const float*)d_in[7];
    const float* du1_b1  = (const float*)d_in[8];
    const float* du1_w2  = (const float*)d_in[9];
    const float* du1_b2  = (const float*)d_in[10];
    const float* w_h2a   = (const float*)d_in[11];
    const float* b_h2a   = (const float*)d_in[12];
    const float* w_h2b   = (const float*)d_in[13];
    const float* b_h2b   = (const float*)d_in[14];
    const float* du2_w1  = (const float*)d_in[15];
    const float* du2_b1  = (const float*)d_in[16];
    const float* du2_w2  = (const float*)d_in[17];
    const float* du2_b2  = (const float*)d_in[18];
    const float* w_tail  = (const float*)d_in[19];
    const float* b_tail  = (const float*)d_in[20];
    const float* basis   = (const float*)d_in[21];

    // workspace layout (bufB region is reused for wgt afterwards)
    unsigned char* ws = (unsigned char*)d_ws;
    size_t off = 0;
    auto alloc = [&](size_t bytes) -> void* {
        void* p = ws + off;
        off = (off + bytes + 255) & ~(size_t)255;
        return p;
    };
    bf16_t* bufA   = (bf16_t*)alloc(FE * 2);             // 33.5 MB
    float*  wgt    = (float*)alloc(NPIX * 72 * 4);       // 37.7 MB (also bufB)
    bf16_t* bufB   = (bf16_t*)wgt;                       // dead once tail runs
    bf16_t* wpA1   = (bf16_t*)alloc((size_t)128 * 1152 * 2);
    bf16_t* wpB1   = (bf16_t*)alloc((size_t)128 * 1152 * 2);
    bf16_t* wpA2   = (bf16_t*)alloc((size_t)128 * 1152 * 2);
    bf16_t* wpB2   = (bf16_t*)alloc((size_t)128 * 1152 * 2);
    bf16_t* wpT    = (bf16_t*)alloc((size_t)80 * 1152 * 2);
    float*  pooled = (float*)alloc(1024 * 4);
    float*  scale  = (float*)alloc(1024 * 4);

    // pack weights for the WMMA convs
    pack_w_kernel<<<cdiv_i(128LL * 1152, 256), 256, 0, stream>>>(w_h1a, wpA1, 128, 128, 128);
    pack_w_kernel<<<cdiv_i(128LL * 1152, 256), 256, 0, stream>>>(w_h1b, wpB1, 128, 128, 128);
    pack_w_kernel<<<cdiv_i(128LL * 1152, 256), 256, 0, stream>>>(w_h2a, wpA2, 128, 128, 128);
    pack_w_kernel<<<cdiv_i(128LL * 1152, 256), 256, 0, stream>>>(w_h2b, wpB2, 128, 128, 128);
    pack_w_kernel<<<cdiv_i(80LL * 1152, 256), 256, 0, stream>>>(w_tail, wpT, 72, 80, 128);

    // trunk
    head_conv_kernel<<<cdiv_i(FE, 256), 256, 0, stream>>>(x, w_head, b_head, bufA, B, H, W);

    const int convGrid = (int)(NPIX / MTILE);            // 4096 tiles
    conv3x3_wmma_kernel<<<convGrid, 256, 0, stream>>>(bufA, wpA1, b_h1a, bufB, nullptr, H, W, 1, 128, 128);
    conv3x3_wmma_kernel<<<convGrid, 256, 0, stream>>>(bufB, wpB1, b_h1b, bufA, nullptr, H, W, 1, 128, 128);

    se_pool_kernel<<<B * 128, 256, 0, stream>>>(bufA, pooled, HW);
    se_mlp_kernel<<<B, 128, 0, stream>>>(pooled, du1_w1, du1_b1, du1_w2, du1_b2, scale, S);
    se_scale_kernel<<<cdiv_i(FE, 256), 256, 0, stream>>>(bufA, scale, HW, FE);

    conv3x3_wmma_kernel<<<convGrid, 256, 0, stream>>>(bufA, wpA2, b_h2a, bufB, nullptr, H, W, 1, 128, 128);
    conv3x3_wmma_kernel<<<convGrid, 256, 0, stream>>>(bufB, wpB2, b_h2b, bufA, nullptr, H, W, 1, 128, 128);

    se_pool_kernel<<<B * 128, 256, 0, stream>>>(bufA, pooled, HW);
    se_mlp_kernel<<<B, 128, 0, stream>>>(pooled, du2_w1, du2_b1, du2_w2, du2_b2, scale, S);
    se_scale_kernel<<<cdiv_i(FE, 256), 256, 0, stream>>>(bufA, scale, HW, FE);

    // tail conv (72 out channels padded to 80 -> 5 o-tiles, f32 NHWK output)
    conv3x3_wmma_kernel<<<convGrid, 5 * 32, 0, stream>>>(bufA, wpT, b_tail, nullptr, wgt, H, W, 0, 72, 72);

    // dynamic filtering into d_out (NCHW f32)
    combine_kernel<<<(int)(NPIX / 256), 256, 0, stream>>>(x, wgt, basis, (float*)d_out, B, H, W);
}
